// CaptionDecoder_77558519431877
// MI455X (gfx1250) — matrix operand order repaired and checked by reading, not compiled, not run on hardware
//
#include <hip/hip_runtime.h>

// ---------------------------------------------------------------------------
// CDNA5 (gfx1250) LSTM caption decoder.
//   B=32, T=63 steps, E=H=512, V=32000, 2 layers.
//   GEMMs via v_wmma_f32_16x16x32_bf16 (fp32 accumulate).
//   Logits GEMM streams its B operand through the Tensor Data Mover
//   (tensor_load_to_lds, TENSORcnt) with double-buffered LDS slabs.
// ---------------------------------------------------------------------------

typedef __bf16 bf16t;
typedef __attribute__((ext_vector_type(16))) __bf16 v16bf;
typedef __attribute__((ext_vector_type(8)))  __bf16 v8bf;
typedef __attribute__((ext_vector_type(8)))  float  v8f;
typedef __attribute__((ext_vector_type(4)))  unsigned int uint4v;
typedef __attribute__((ext_vector_type(8)))  int int8v;
typedef __attribute__((ext_vector_type(4)))  int int4v;

#define BATCH   32
#define TSTEPS  63
#define HID     512
#define EMB     512
#define KCOMB   1024
#define VOCAB   32000

// ---- WMMA operand loaders (CDNA5 ISA §7.12.2 layouts) -----------------

// A-matrix 16x32 bf16 (MxK), row-major source, lda in elements.
// lane<16:  holds M=lane,    K = {0..7, 16..23}
// lane>=16: holds M=lane-16, K = {8..15, 24..31}
__device__ __forceinline__ v16bf load_a16x32(const bf16t* __restrict__ base,
                                             int lda, int lane) {
    const int m  = lane & 15;
    const int k0 = (lane >> 4) << 3;            // 0 or 8
    const bf16t* p = base + (size_t)m * lda + k0;
    v8bf lo = *(const v8bf*)(p);                // K = k0 + 0..7
    v8bf hi = *(const v8bf*)(p + 16);           // K = k0 + 16..23
    return __builtin_shufflevector(lo, hi, 0,1,2,3,4,5,6,7,8,9,10,11,12,13,14,15);
}

// B-matrix 32x16 bf16 (KxN). Source is W stored row-major [N][K], so column n
// of B == row n of W: lane holds two 8-element K runs of W[n] (kb..kb+7, kb+16..kb+23).
// lane<16: N=lane, K={0..7,16..23} ; lane>=16: N=lane-16, K={8..15,24..31}
// Works for both global and LDS sources (addrspace inferred -> ds_load_b128).
__device__ __forceinline__ v16bf load_b32x16(const bf16t* wrow0,
                                             size_t ldw, int lane) {
    const int n  = lane & 15;
    const int kb = (lane >> 4) << 3;            // 0 or 8
    const bf16t* p = wrow0 + (size_t)n * ldw + kb;
    v8bf lo = *(const v8bf*)(p);
    v8bf hi = *(const v8bf*)(p + 16);
    return __builtin_shufflevector(lo, hi, 0,1,2,3,4,5,6,7,8,9,10,11,12,13,14,15);
}

__device__ __forceinline__ v8f wmma_bf16(v16bf a, v16bf b, v8f c) {
    return __builtin_amdgcn_wmma_f32_16x16x32_bf16(false, a, false, b,
                                                   (short)0, c, false, false);
}

__device__ __forceinline__ float sigmoidf_fast(float x) {
    return 1.0f / (1.0f + __expf(-x));
}

// ---------------------------------------------------------------------------
// Tensor Data Mover: 2-D bf16 tile load Global -> LDS (CDNA5 ISA
// §8.3/§8.4).  Issued by one wave; tracked with TENSORcnt; EXEC ignored.
//   tile_d0 contiguous K elements per row, tile_d1 rows, row stride in elems.
//   pad codes: after every 2^(pi+1) DWORDs insert (pa+1) DWORDs in LDS
//   (pa < 0 disables padding).
// ---------------------------------------------------------------------------
__device__ __forceinline__ void tdm_load_2d(const void* lds_ptr,
                                            const bf16t* gptr,
                                            unsigned tile_d0, unsigned tile_d1,
                                            unsigned long long stride_elems,
                                            int pi, int pa) {
    unsigned long long ga = (unsigned long long)(uintptr_t)gptr;
    // Flat LDS pointers keep the LDS byte offset in the low 32 bits
    unsigned int lds = (unsigned int)(uintptr_t)lds_ptr;

    uint4v g0;
    g0.x = 1u;                                          // count=1, user mode
    g0.y = lds;                                         // lds_addr
    g0.z = (unsigned int)ga;                            // global_addr[31:0]
    g0.w = ((unsigned int)(ga >> 32) & 0x01FFFFFFu)     // global_addr[56:32]
         | (2u << 30);                                  // type = 2 ("image")

    const unsigned int td0 = 1u << 30, td1 = 1u << 30;  // tensor dims (no OOB)
    unsigned int w0 = (1u << 16);                       // data_size = 2 bytes
    if (pa >= 0) {
        w0 |= (1u << 20)                                // pad_enable
            | ((unsigned int)pi << 22)                  // pad_interval
            | ((unsigned int)pa << 25);                 // pad_amount
    }
    int8v g1;
    g1[0] = (int)w0;                                    // wg_mask=0
    g1[1] = (int)((td0 & 0xFFFFu) << 16);               // tensor_dim0[15:0]
    g1[2] = (int)(((td0 >> 16) & 0xFFFFu) | ((td1 & 0xFFFFu) << 16));
    g1[3] = (int)(((td1 >> 16) & 0xFFFFu) | (tile_d0 << 16));
    g1[4] = (int)(tile_d1 & 0xFFFFu);                   // tile_dim2 = 0
    g1[5] = (int)(unsigned int)stride_elems;            // dim0_stride[31:0]
    g1[6] = (int)((unsigned int)(stride_elems >> 32) & 0xFFFFu);
    g1[7] = 0;                                          // dim1_stride (2-D)

    int4v z4 = {0, 0, 0, 0};                            // groups 2/3 unused
    int8v z8 = {0, 0, 0, 0, 0, 0, 0, 0};                // 6-arg toolchain extra
    __builtin_amdgcn_tensor_load_to_lds(g0, g1, z4, z4, z8, 0);
}

// ---------------------------------------------------------------------------
// Prep kernels
// ---------------------------------------------------------------------------

// Gate weights: src [2][512][1024] f32 -> dst[(l*4+gate)][512][1024] bf16
__global__ void cvt_gate_kernel(const float* __restrict__ src,
                                bf16t* __restrict__ dst, int gate) {
    const size_t S = (size_t)HID * KCOMB;
    size_t i = (size_t)blockIdx.x * 256 + threadIdx.x;   // over 2*S
    size_t l = i / S, r = i - l * S;
    dst[(l * 4 + gate) * S + r] = (bf16t)src[i];
}

// Flat f32 -> bf16 convert (out_w)
__global__ void cvt_flat_kernel(const float* __restrict__ src,
                                bf16t* __restrict__ dst) {
    size_t i = (size_t)blockIdx.x * 256 + threadIdx.x;
    dst[i] = (bf16t)src[i];
}

// Embedding gather: emb[t][b][k] = bf16(embedding_w[captions[b][t]][k]), t<63
__global__ void embed_kernel(const int* __restrict__ captions,
                             const float* __restrict__ embw,
                             bf16t* __restrict__ emb) {
    size_t i = (size_t)blockIdx.x * 256 + threadIdx.x;   // 63*32*512
    int k = (int)(i & (EMB - 1));
    int b = (int)((i >> 9) & 31);
    int t = (int)(i >> 14);
    int tok = captions[b * 64 + t];                      // captions[:, :-1]
    emb[i] = (bf16t)embw[(size_t)tok * EMB + k];
}

// Initial state: h0 = feat @ init_h_w.T + b ; c0 likewise; broadcast to layers.
__global__ void init_state_kernel(const float* __restrict__ feat,
                                  const float* __restrict__ ihw,
                                  const float* __restrict__ ihb,
                                  const float* __restrict__ icw,
                                  const float* __restrict__ icb,
                                  float* __restrict__ cbl,
                                  bf16t* __restrict__ hbf) {
    int idx = blockIdx.x * 256 + threadIdx.x;            // 32*512
    int b = idx >> 9, n = idx & (HID - 1);
    const float* f  = feat + (size_t)b * EMB;
    const float* wh = ihw  + (size_t)n * EMB;
    const float* wc = icw  + (size_t)n * EMB;
    float ah = ihb[n], ac = icb[n];
    for (int k = 0; k < EMB; ++k) {
        float x = f[k];
        ah = fmaf(x, wh[k], ah);
        ac = fmaf(x, wc[k], ac);
    }
    cbl[(size_t)(0 * BATCH + b) * HID + n] = ac;
    cbl[(size_t)(1 * BATCH + b) * HID + n] = ac;
    bf16t hb = (bf16t)ah;
    hbf[((size_t)(0 * 2 + 0) * BATCH + b) * HID + n] = hb;  // parity0, layer0
    hbf[((size_t)(0 * 2 + 1) * BATCH + b) * HID + n] = hb;  // parity0, layer1
}

// ---------------------------------------------------------------------------
// One LSTM (t, layer) step.  grid = 8 blocks (64-wide hidden slices),
// 256 threads = 8 waves.  Wave w: gate g=w/2, 32-wide N half = (w&1)*32.
// Each wave: 2x2 tiles of 16x16, K loop over comb = [x | h_prev] (1024).
// Epilogue: gates staged in LDS, elementwise LSTM update owned locally.
// ---------------------------------------------------------------------------
__global__ __launch_bounds__(256)
void lstm_step_kernel(const bf16t* __restrict__ xsrc,    // [32][512] bf16
                      const bf16t* __restrict__ hprev,   // [32][512] bf16
                      const bf16t* __restrict__ wgl,     // gates f,i,c,o: [4][512][1024]
                      const float* __restrict__ bfv, const float* __restrict__ biv,
                      const float* __restrict__ bcv, const float* __restrict__ bov,
                      float* __restrict__ cbl,           // [32][512] f32 (this layer)
                      bf16t* __restrict__ hout,          // [32][512] bf16
                      bf16t* __restrict__ hall_t) {      // [32][512] bf16 or null
    __shared__ float gs[4][BATCH][64];

    const int tid  = threadIdx.x;
    const int wv   = tid >> 5;
    const int lane = tid & 31;
    const int g    = wv >> 1;
    const int nsub = (wv & 1) << 5;
    const int n0   = blockIdx.x << 6;

    const size_t GS = (size_t)HID * KCOMB;
    const bf16t* wbase = wgl + (size_t)g * GS;

    v8f acc00 = {}, acc01 = {}, acc10 = {}, acc11 = {};

    for (int kk = 0; kk < KCOMB; kk += 32) {
        const bf16t* asrc; int kc;
        if (kk < EMB) { asrc = xsrc;  kc = kk; }
        else          { asrc = hprev; kc = kk - EMB; }

        v16bf a0 = load_a16x32(asrc + kc,            EMB, lane);
        v16bf a1 = load_a16x32(asrc + 16 * EMB + kc, EMB, lane);
        v16bf b0 = load_b32x16(wbase + (size_t)(n0 + nsub)      * KCOMB + kk, KCOMB, lane);
        v16bf b1 = load_b32x16(wbase + (size_t)(n0 + nsub + 16) * KCOMB + kk, KCOMB, lane);

        acc00 = wmma_bf16(a0, b0, acc00);
        acc01 = wmma_bf16(a0, b1, acc01);
        acc10 = wmma_bf16(a1, b0, acc10);
        acc11 = wmma_bf16(a1, b1, acc11);
    }

    // C-tile layout: element v -> (M = v + 8*(lane>=16), N = lane&15)
    const int crow = (lane >> 4) << 3;
    const int ccol = lane & 15;
    for (int v = 0; v < 8; ++v) {
        gs[g][     crow + v][nsub +      ccol] = acc00[v];
        gs[g][     crow + v][nsub + 16 + ccol] = acc01[v];
        gs[g][16 + crow + v][nsub +      ccol] = acc10[v];
        gs[g][16 + crow + v][nsub + 16 + ccol] = acc11[v];
    }
    __syncthreads();

    // Elementwise LSTM update: 32x64 = 2048 elems, 8 per thread.
    for (int r = 0; r < 8; ++r) {
        int e  = r * 256 + tid;
        int m  = e >> 6;
        int nl = e & 63;
        int n  = n0 + nl;
        float fg = sigmoidf_fast(gs[0][m][nl] + bfv[n]);
        float ig = sigmoidf_fast(gs[1][m][nl] + biv[n]);
        float cg = tanhf        (gs[2][m][nl] + bcv[n]);
        float og = sigmoidf_fast(gs[3][m][nl] + bov[n]);
        size_t ci = (size_t)m * HID + n;
        float cn = fg * cbl[ci] + ig * cg;
        cbl[ci] = cn;
        float h = og * tanhf(cn);
        bf16t hb = (bf16t)h;
        hout[ci] = hb;
        if (hall_t) hall_t[ci] = hb;
    }
}

// ---------------------------------------------------------------------------
// Batched logits GEMM: C[2016][32000] = A[2016][512] @ W[32000][512]^T + bias.
// Row m = t*32 + b, output permuted to out[b][t][v].
// Workgroup tile: M=32, N=256 (8 waves x 32N).  K loop over 512 in 32-slabs.
// B slab (256 rows x 32 K bf16 = 16 KB) is streamed Global->LDS by the
// Tensor Data Mover, double-buffered; TDM block-padding (4 DW after every
// 16 DW) gives an 80-byte LDS row pitch to spread bank accesses.
// ---------------------------------------------------------------------------
#define BROW 40                       // LDS row pitch in bf16 elems (32 + 8 pad)

__global__ __launch_bounds__(256)
void logits_gemm_kernel(const bf16t* __restrict__ A,
                        const bf16t* __restrict__ W,
                        const float* __restrict__ bias,
                        float* __restrict__ out) {
    __shared__ __align__(16) bf16t bbuf[2][256 * BROW];   // 2 x 20 KB

    const int tid  = threadIdx.x;
    const int wv   = tid >> 5;
    const int lane = tid & 31;
    const int m0   = blockIdx.y << 5;
    const int n0   = blockIdx.x << 8;
    const int nw   = wv << 5;                 // this wave's N offset in slab

    // Prime the pipeline: slab for kk=0 into buffer 0.
    if (wv == 0) {
        tdm_load_2d(&bbuf[0][0], W + (size_t)n0 * HID, 32, 256, HID,
                    /*pad: 16 DW interval*/3, /*4 DW amount*/3);
    }

    v8f acc00 = {}, acc01 = {}, acc10 = {}, acc11 = {};

    for (int i = 0; i < HID / 32; ++i) {
        const int kk = i << 5;
        if (wv == 0) {
            if (i + 1 < HID / 32) {           // issue-ahead next slab
                tdm_load_2d(&bbuf[(i + 1) & 1][0],
                            W + (size_t)n0 * HID + (kk + 32), 32, 256, HID, 3, 3);
                __builtin_amdgcn_s_wait_tensorcnt((short)1);  // current done
            } else {
                __builtin_amdgcn_s_wait_tensorcnt((short)0);
            }
        }
        __syncthreads();                      // slab visible to all waves

        const bf16t* bb = &bbuf[i & 1][0];
        v16bf a0 = load_a16x32(A + (size_t)m0        * HID + kk, HID, lane);
        v16bf a1 = load_a16x32(A + (size_t)(m0 + 16) * HID + kk, HID, lane);
        v16bf b0 = load_b32x16(bb + (size_t)nw        * BROW, BROW, lane);
        v16bf b1 = load_b32x16(bb + (size_t)(nw + 16) * BROW, BROW, lane);

        acc00 = wmma_bf16(a0, b0, acc00);
        acc01 = wmma_bf16(a0, b1, acc01);
        acc10 = wmma_bf16(a1, b0, acc10);
        acc11 = wmma_bf16(a1, b1, acc11);

        __syncthreads();                      // all reads done before slab reuse
    }

    const int crow = (lane >> 4) << 3;
    const int ccol = lane & 15;
    #pragma unroll
    for (int mt = 0; mt < 2; ++mt) {
        #pragma unroll
        for (int nt = 0; nt < 2; ++nt) {
            const v8f acc = (mt == 0) ? (nt == 0 ? acc00 : acc01)
                                      : (nt == 0 ? acc10 : acc11);
            int n  = n0 + nw + nt * 16 + ccol;
            float bv = bias[n];
            #pragma unroll
            for (int v = 0; v < 8; ++v) {
                int m = m0 + mt * 16 + crow + v;
                int t = m >> 5;          // row = t*32 + b
                int b = m & 31;
                out[((size_t)b * TSTEPS + t) * VOCAB + n] = acc[v] + bv;
            }
        }
    }
}

// ---------------------------------------------------------------------------
// Host launcher
// ---------------------------------------------------------------------------
extern "C" void kernel_launch(void* const* d_in, const int* in_sizes, int n_in,
                              void* d_out, int out_size, void* d_ws, size_t ws_size,
                              hipStream_t stream) {
    const float* features = (const float*)d_in[0];
    const int*   captions = (const int*)  d_in[1];
    const float* embw     = (const float*)d_in[2];
    const float* Wf       = (const float*)d_in[3];
    const float* bfv      = (const float*)d_in[4];
    const float* Wi       = (const float*)d_in[5];
    const float* biv      = (const float*)d_in[6];
    const float* Wc       = (const float*)d_in[7];
    const float* bcv      = (const float*)d_in[8];
    const float* Wo       = (const float*)d_in[9];
    const float* bov      = (const float*)d_in[10];
    const float* out_w    = (const float*)d_in[11];
    const float* out_b    = (const float*)d_in[12];
    const float* ihw      = (const float*)d_in[13];
    const float* ihb      = (const float*)d_in[14];
    const float* icw      = (const float*)d_in[15];
    const float* icb      = (const float*)d_in[16];
    float* out = (float*)d_out;

    // Workspace layout (all offsets 16B-aligned).
    const size_t S_WG   = 2ull * 4 * HID * KCOMB;        // gate weights bf16
    const size_t S_OUTW = (size_t)VOCAB * HID;           // out_w bf16
    const size_t S_EMB  = (size_t)TSTEPS * BATCH * EMB;  // embeddings bf16
    const size_t S_HBF  = 2ull * 2 * BATCH * HID;        // h (parity x layer) bf16
    const size_t S_HALL = (size_t)TSTEPS * BATCH * HID;  // top-layer h bf16

    bf16t* wg   = (bf16t*)d_ws;
    bf16t* outw = wg   + S_WG;
    bf16t* emb  = outw + S_OUTW;
    bf16t* hbf  = emb  + S_EMB;
    bf16t* hall = hbf  + S_HBF;
    float* cbl  = (float*)(hall + S_HALL);
    (void)ws_size; (void)n_in; (void)in_sizes; (void)out_size;

    const size_t GS = (size_t)HID * KCOMB;               // per-layer gate elems

    // 1) Weight / embedding prep.
    cvt_gate_kernel<<<(unsigned)(2 * GS / 256), 256, 0, stream>>>(Wf, wg, 0);
    cvt_gate_kernel<<<(unsigned)(2 * GS / 256), 256, 0, stream>>>(Wi, wg, 1);
    cvt_gate_kernel<<<(unsigned)(2 * GS / 256), 256, 0, stream>>>(Wc, wg, 2);
    cvt_gate_kernel<<<(unsigned)(2 * GS / 256), 256, 0, stream>>>(Wo, wg, 3);
    cvt_flat_kernel<<<(unsigned)(S_OUTW / 256), 256, 0, stream>>>(out_w, outw);
    embed_kernel<<<(unsigned)(S_EMB / 256), 256, 0, stream>>>(captions, embw, emb);

    // 2) Initial h/c (broadcast to both layers, parity 0).
    init_state_kernel<<<(BATCH * HID) / 256, 256, 0, stream>>>(
        features, ihw, ihb, icw, icb, cbl, hbf);

    // 3) Sequential recurrence: 63 steps x 2 layers. h double-buffered by
    //    t-parity so the recurrent bf16 operand is never read+written by
    //    concurrent workgroups of the same launch.
    for (int t = 0; t < TSTEPS; ++t) {
        const int p = t & 1, q = (t + 1) & 1;
        const size_t L = (size_t)BATCH * HID;

        const bf16t* x0  = emb + (size_t)t * L;
        const bf16t* hp0 = hbf + ((size_t)p * 2 + 0) * L;
        bf16t*       ho0 = hbf + ((size_t)q * 2 + 0) * L;
        lstm_step_kernel<<<HID / 64, 256, 0, stream>>>(
            x0, hp0, wg, bfv, biv, bcv, bov, cbl, ho0, nullptr);

        const bf16t* hp1 = hbf + ((size_t)p * 2 + 1) * L;
        bf16t*       ho1 = hbf + ((size_t)q * 2 + 1) * L;
        lstm_step_kernel<<<HID / 64, 256, 0, stream>>>(
            ho0, hp1, wg + 4 * GS, bfv + HID, biv + HID, bcv + HID, bov + HID,
            cbl + L, ho1, hall + (size_t)t * L);
    }

    // 4) One big batched logits GEMM: 2016 x 32000 x 512 (+bias, permuted out),
    //    B operand streamed via the Tensor Data Mover.
    logits_gemm_kernel<<<dim3(VOCAB / 256, (TSTEPS * BATCH) / 32), 256, 0, stream>>>(
        hall, outw, out_b, out);
}
